// SelfAttention_8495445311572
// MI455X (gfx1250) — matrix operand order repaired and checked by reading, not compiled, not run on hardware
//
#include <hip/hip_runtime.h>

// ---------------------------------------------------------------------------
// Causal self-attention forward for MI455X (gfx1250, wave32, WMMA).
//   B=2, T=2048, C=1024, H=16, D=64.
// bf16 V_WMMA_F32_16X16X32_BF16 with f32 accumulation everywhere.
// - GEMMs: 64x64 register block per wave, double-buffered K pipeline
//   (16 WMMAs per stage overlap the next stage's fragment loads).
// - Fused attention keeps the 16x2048 f32 score panel in LDS (132KB of the
//   320KB/WGP CDNA5 LDS); att_weights (537MB) is written once with NT stores.
// ---------------------------------------------------------------------------

#define DEV __device__ __forceinline__

typedef __attribute__((ext_vector_type(16))) __bf16 v16bf;
typedef __attribute__((ext_vector_type(8)))  __bf16 v8bf;
typedef __attribute__((ext_vector_type(8)))  float  v8f;
typedef __attribute__((ext_vector_type(4)))  float  v4f;

constexpr int Bb = 2, Tt = 2048, Cc = 1024, Hh = 16, Dd = 64;
constexpr int Mrows = Bb * Tt;   // 4096
constexpr int N3 = 3 * Cc;       // 3072
constexpr int LDSS = 2064;       // padded LDS row stride (floats)

DEV __bf16 f2bf(float f) { return (__bf16)f; }

DEV v8f wmma_bf16(v16bf a, v16bf b, v8f c) {
  return __builtin_amdgcn_wmma_f32_16x16x32_bf16(false, a, false, b, (short)0, c,
                                                 false, false);
}

// A-matrix (16x32 bf16) fragment, row-major source.
DEV v16bf load_A_rowmajor(const __bf16* base, int ld, int kb, int lane) {
  int m = lane & 15, h = (lane >> 4) & 1;
  const __bf16* p0 = base + (size_t)m * ld + kb + h * 8;
  v8bf a0 = *(const v8bf*)p0;
  v8bf a1 = *(const v8bf*)(p0 + 16);
  v16bf r;
#pragma unroll
  for (int i = 0; i < 8; ++i) { r[i] = a0[i]; r[i + 8] = a1[i]; }
  return r;
}

// B-matrix (32x16 bf16) fragment from bt[n][k] row-major ("B-transposed").
DEV v16bf load_B_colmajor(const __bf16* bt, int ld, int nb, int kb, int lane) {
  int n = lane & 15, h = (lane >> 4) & 1;
  const __bf16* p0 = bt + (size_t)(nb + n) * ld + kb + h * 8;
  v8bf b0 = *(const v8bf*)p0;
  v8bf b1 = *(const v8bf*)(p0 + 16);
  v16bf r;
#pragma unroll
  for (int i = 0; i < 8; ++i) { r[i] = b0[i]; r[i + 8] = b1[i]; }
  return r;
}

// ---------------------------------------------------------------- converters
__global__ void cvt_f32_to_bf16(const float* __restrict__ src,
                                __bf16* __restrict__ dst, int n) {
  int i = blockIdx.x * blockDim.x + threadIdx.x;
  if (i < n) dst[i] = f2bf(src[i]);
}

// wt[n*K + k] = w[k*N + n]
__global__ void transpose_f32_to_bf16(const float* __restrict__ w,
                                      __bf16* __restrict__ wt, int K, int N) {
  int i = blockIdx.x * blockDim.x + threadIdx.x;
  if (i < K * N) {
    int k = i / N, n = i % N;
    wt[(size_t)n * K + k] = f2bf(w[i]);
  }
}

// ------------------------------------------------------------------ QKV GEMM
__global__ void __launch_bounds__(128)
qkv_gemm_kernel(const __bf16* __restrict__ xbf, const __bf16* __restrict__ waT,
                const float* __restrict__ b_attn, __bf16* __restrict__ q,
                __bf16* __restrict__ k, __bf16* __restrict__ vT) {
  int lane = threadIdx.x & 31;
  int wblk = blockIdx.x * 4 + (threadIdx.x >> 5);
  constexpr int NB = N3 / 64;                  // 48 column blocks
  int mt = (wblk / NB) * 64;
  int nt = (wblk % NB) * 64;
  int m16 = lane & 15, h8 = lane >> 4;

  v8f acc[4][4] = {};
  v16bf aF0[4], bF0[4], aF1[4], bF1[4];

  auto loadA = [&](v16bf* dst, int kb) {
#pragma unroll
    for (int i = 0; i < 4; ++i)
      dst[i] = load_A_rowmajor(xbf + (size_t)(mt + i * 16) * Cc, Cc, kb, lane);
  };
  auto loadB = [&](v16bf* dst, int kb) {
#pragma unroll
    for (int j = 0; j < 4; ++j)
      dst[j] = load_B_colmajor(waT, Cc, nt + j * 16, kb, lane);
  };

  loadA(aF0, 0);
  loadB(bF0, 0);
  for (int kb = 0; kb < Cc; kb += 64) {        // double-buffered 2x32 K steps
    loadA(aF1, kb + 32);
    loadB(bF1, kb + 32);
    __builtin_prefetch(xbf + (size_t)(mt + m16) * Cc + kb + 128, 0, 1);
#pragma unroll
    for (int i = 0; i < 4; ++i)
#pragma unroll
      for (int j = 0; j < 4; ++j)
        acc[i][j] = wmma_bf16(aF0[i], bF0[j], acc[i][j]);
    if (kb + 64 < Cc) { loadA(aF0, kb + 64); loadB(bF0, kb + 64); }
#pragma unroll
    for (int i = 0; i < 4; ++i)
#pragma unroll
      for (int j = 0; j < 4; ++j)
        acc[i][j] = wmma_bf16(aF1[i], bF1[j], acc[i][j]);
  }

  // Scatter epilogue. part / head / batch are wave-uniform (64-wide blocks
  // never straddle q/k/v, head, or batch boundaries) -> scalar branches.
  int part = nt >> 10;                         // 0=q 1=k 2=v (uniform)
  int hh = (nt & (Cc - 1)) >> 6;               // head (uniform)
  int bbat = mt >> 11;                         // batch (uniform)
  int mloc = mt & (Tt - 1);                    // t of tile row 0 (uniform)
  int bh = bbat * Hh + hh;
  if (part < 2) {
    __bf16* dstb = (part == 0 ? q : k) + ((size_t)bh * Tt + mloc) * Dd;
#pragma unroll
    for (int j = 0; j < 4; ++j) {
      int dd = j * 16 + m16;
      float bias = b_attn[nt + dd];
      __bf16* dst = dstb + dd;
#pragma unroll
      for (int i = 0; i < 4; ++i)
#pragma unroll
        for (int r = 0; r < 8; ++r)
          dst[(size_t)(i * 16 + r + 8 * h8) * Dd] = f2bf(acc[i][j][r] + bias);
    }
  } else {
    // vT[bh][d][t]: the 8 accumulator rows are 8 consecutive t -> v8bf store
#pragma unroll
    for (int j = 0; j < 4; ++j) {
      int dd = j * 16 + m16;
      float bias = b_attn[nt + dd];
      __bf16* dst = vT + ((size_t)bh * Dd + dd) * Tt + mloc + 8 * h8;
#pragma unroll
      for (int i = 0; i < 4; ++i) {
        v8bf pk;
#pragma unroll
        for (int r = 0; r < 8; ++r) pk[r] = f2bf(acc[i][j][r] + bias);
        *(v8bf*)(dst + i * 16) = pk;
      }
    }
  }
}

// -------------------------------------------------------- fused attention
__global__ void __launch_bounds__(128)
attn_kernel(const __bf16* __restrict__ q, const __bf16* __restrict__ k,
            const __bf16* __restrict__ vT, float* __restrict__ att,
            __bf16* __restrict__ yh) {
  extern __shared__ float sc[];                // 16 * LDSS floats
  __shared__ float redbuf[16 * 8];
  __shared__ float rowstat[16];

  int tid = threadIdx.x;
  int lane = tid & 31;
  int w = tid >> 5;
  int bh = blockIdx.x >> 7;
  int qtile = blockIdx.x & 127;
  int qt0 = qtile * 16;
  int ncols = qt0 + 16;                        // causal: valid key columns
  int m16 = lane & 15, h8 = lane >> 4;

  const __bf16* qh = q + ((size_t)bh * Tt + qt0) * Dd;
  v16bf aq0 = load_A_rowmajor(qh, Dd, 0, lane);
  v16bf aq1 = load_A_rowmajor(qh, Dd, 32, lane);

  // ---- scores: waves split key tiles; q fragments reused from registers
  for (int kt = w; kt <= qtile; kt += 4) {
    const __bf16* kh = k + ((size_t)bh * Tt + kt * 16) * Dd;
    v16bf b0 = load_B_colmajor(kh, Dd, 0, 0, lane);
    v16bf b1 = load_B_colmajor(kh, Dd, 0, 32, lane);
    v8f acc = {};
    acc = wmma_bf16(aq0, b0, acc);
    acc = wmma_bf16(aq1, b1, acc);
#pragma unroll
    for (int r = 0; r < 8; ++r) {
      int mm = r + 8 * h8;
      int col = kt * 16 + m16;
      float s = acc[r] * 0.125f;               // 1/sqrt(64)
      if (col > qt0 + mm) s = -1e30f;          // causal mask
      sc[mm * LDSS + col] = s;
    }
  }
  __syncthreads();

  // ---- softmax: 8 threads per row
  int row = tid >> 3, sub = tid & 7;
  float mx = -3.0e38f;
  for (int c = sub; c < ncols; c += 8) mx = fmaxf(mx, sc[row * LDSS + c]);
  redbuf[row * 8 + sub] = mx;
  __syncthreads();
  if (sub == 0) {
    float v = redbuf[row * 8];
#pragma unroll
    for (int i = 1; i < 8; ++i) v = fmaxf(v, redbuf[row * 8 + i]);
    rowstat[row] = v;
  }
  __syncthreads();
  float rm = rowstat[row];
  float sum = 0.f;
  for (int c = sub; c < ncols; c += 8) {
    float e = __expf(sc[row * LDSS + c] - rm);
    sc[row * LDSS + c] = e;
    sum += e;
  }
  redbuf[row * 8 + sub] = sum;
  __syncthreads();
  if (sub == 0) {
    float v = 0.f;
#pragma unroll
    for (int i = 0; i < 8; ++i) v += redbuf[row * 8 + i];
    rowstat[row] = 1.0f / v;
  }
  __syncthreads();
  float inv = rowstat[row];
  for (int c = sub; c < ncols; c += 8) sc[row * LDSS + c] *= inv;
  __syncthreads();

  // ---- write probabilities once, non-temporal (537MB streaming output)
  float* arow = att + ((size_t)bh * Tt + qt0) * Tt;
#pragma unroll
  for (int rr = 0; rr < 4; ++rr) {
    int mm = w * 4 + rr;
    float* dst = arow + (size_t)mm * Tt;
    for (int c = lane * 4; c < Tt; c += 128) {
      v4f val;
#pragma unroll
      for (int i = 0; i < 4; ++i)
        val[i] = (c + i < ncols) ? sc[mm * LDSS + c + i] : 0.0f;
      __builtin_nontemporal_store(val, (v4f*)(dst + c));
    }
  }

  // ---- y = P @ V : wave w owns output columns [16w, 16w+16)
  int dg = w;
  const __bf16* vh = vT + (size_t)bh * Dd * Tt;
  v8f acc = {};
  int nfull = ncols & ~31;                     // guard-free 32-wide steps
  for (int kt = 0; kt < nfull; kt += 32) {
    const float* srow = &sc[m16 * LDSS + kt + h8 * 8];
    v4f f0 = *(const v4f*)(srow);
    v4f f1 = *(const v4f*)(srow + 4);
    v4f f2 = *(const v4f*)(srow + 16);
    v4f f3 = *(const v4f*)(srow + 20);
    v16bf a;
#pragma unroll
    for (int i = 0; i < 4; ++i) {
      a[i]      = f2bf(f0[i]);
      a[4 + i]  = f2bf(f1[i]);
      a[8 + i]  = f2bf(f2[i]);
      a[12 + i] = f2bf(f3[i]);
    }
    v16bf b = load_B_colmajor(vh, Tt, dg * 16, kt, lane);
    acc = wmma_bf16(a, b, acc);
  }
  if (ncols & 16) {                            // 16-wide tail
    int kt = nfull;
    const float* srow = &sc[m16 * LDSS + kt + h8 * 8];
    v4f f0 = *(const v4f*)(srow);
    v4f f1 = *(const v4f*)(srow + 4);
    v16bf a;
#pragma unroll
    for (int i = 0; i < 4; ++i) {
      a[i]      = f2bf(f0[i]);
      a[4 + i]  = f2bf(f1[i]);
      a[8 + i]  = f2bf(0.0f);
      a[12 + i] = f2bf(0.0f);
    }
    v16bf b = load_B_colmajor(vh, Tt, dg * 16, kt, lane);
    acc = wmma_bf16(a, b, acc);
  }
#pragma unroll
  for (int r = 0; r < 8; ++r) {
    int mm = r + 8 * h8;
    yh[((size_t)bh * Tt + qt0 + mm) * Dd + dg * 16 + m16] = f2bf(acc[r]);
  }
}

// ------------------------------------------------------------- output proj
__global__ void __launch_bounds__(128)
proj_gemm_kernel(const __bf16* __restrict__ yh, const __bf16* __restrict__ wpT,
                 const float* __restrict__ b_proj, float* __restrict__ yout) {
  int lane = threadIdx.x & 31;
  int wblk = blockIdx.x * 4 + (threadIdx.x >> 5);
  constexpr int NB = Cc / 64;                  // 16 column blocks
  int mt = (wblk / NB) * 64;
  int nt = (wblk % NB) * 64;
  int m16 = lane & 15, h8 = lane >> 4;
  int bbat = mt >> 11;                         // uniform across the block
  int mloc = mt & (Tt - 1);

  v8f acc[4][4] = {};
  v16bf aF0[4], bF0[4], aF1[4], bF1[4];

  auto loadA = [&](v16bf* dst, int kb) {
    int hd = kb >> 6;
    int d0 = (kb & 63) + h8 * 8;
#pragma unroll
    for (int i = 0; i < 4; ++i) {
      const __bf16* p = yh +
          ((size_t)(bbat * Hh + hd) * Tt + mloc + i * 16 + m16) * Dd + d0;
      v8bf a0 = *(const v8bf*)p;
      v8bf a1 = *(const v8bf*)(p + 16);
      v16bf a;
#pragma unroll
      for (int e = 0; e < 8; ++e) { a[e] = a0[e]; a[e + 8] = a1[e]; }
      dst[i] = a;
    }
  };
  auto loadB = [&](v16bf* dst, int kb) {
#pragma unroll
    for (int j = 0; j < 4; ++j)
      dst[j] = load_B_colmajor(wpT, Cc, nt + j * 16, kb, lane);
  };

  loadA(aF0, 0);
  loadB(bF0, 0);
  for (int kb = 0; kb < Cc; kb += 64) {
    loadA(aF1, kb + 32);
    loadB(bF1, kb + 32);
#pragma unroll
    for (int i = 0; i < 4; ++i)
#pragma unroll
      for (int j = 0; j < 4; ++j)
        acc[i][j] = wmma_bf16(aF0[i], bF0[j], acc[i][j]);
    if (kb + 64 < Cc) { loadA(aF0, kb + 64); loadB(bF0, kb + 64); }
#pragma unroll
    for (int i = 0; i < 4; ++i)
#pragma unroll
      for (int j = 0; j < 4; ++j)
        acc[i][j] = wmma_bf16(aF1[i], bF1[j], acc[i][j]);
  }

#pragma unroll
  for (int j = 0; j < 4; ++j) {
    int n = nt + j * 16 + m16;
    float bias = b_proj[n];
#pragma unroll
    for (int i = 0; i < 4; ++i)
#pragma unroll
      for (int r = 0; r < 8; ++r) {
        int rowg = mt + i * 16 + r + 8 * h8;
        yout[(size_t)rowg * Cc + n] = acc[i][j][r] + bias;
      }
  }
}

// ---------------------------------------------------------------------------
extern "C" void kernel_launch(void* const* d_in, const int* in_sizes, int n_in,
                              void* d_out, int out_size, void* d_ws, size_t ws_size,
                              hipStream_t stream) {
  const float* x      = (const float*)d_in[0];
  // d_in[1] = attention_mask (pure causal; applied analytically, not read)
  const float* W_attn = (const float*)d_in[2];
  const float* b_attn = (const float*)d_in[3];
  const float* W_proj = (const float*)d_in[4];
  const float* b_proj = (const float*)d_in[5];

  char* ws = (char*)d_ws;
  size_t off = 0;
  auto alloc = [&](size_t bytes) -> void* {
    void* p = ws + off;
    off = (off + bytes + 255) & ~(size_t)255;
    return p;
  };
  __bf16* xbf = (__bf16*)alloc((size_t)Mrows * Cc * 2);        // 8 MiB
  __bf16* waT = (__bf16*)alloc((size_t)N3 * Cc * 2);           // 6 MiB
  __bf16* wpT = (__bf16*)alloc((size_t)Cc * Cc * 2);           // 2 MiB
  __bf16* qw  = (__bf16*)alloc((size_t)Bb * Hh * Tt * Dd * 2); // 8 MiB
  __bf16* kw  = (__bf16*)alloc((size_t)Bb * Hh * Tt * Dd * 2); // 8 MiB
  __bf16* vT  = (__bf16*)alloc((size_t)Bb * Hh * Tt * Dd * 2); // 8 MiB
  __bf16* yh  = (__bf16*)alloc((size_t)Bb * Hh * Tt * Dd * 2); // 8 MiB
  (void)ws_size; (void)in_sizes; (void)n_in; (void)out_size;

  float* yout = (float*)d_out;
  float* att  = yout + (size_t)Mrows * Cc;

  cvt_f32_to_bf16<<<(Mrows * Cc) / 256, 256, 0, stream>>>(x, xbf, Mrows * Cc);
  transpose_f32_to_bf16<<<(Cc * N3) / 256, 256, 0, stream>>>(W_attn, waT, Cc, N3);
  transpose_f32_to_bf16<<<(Cc * Cc) / 256, 256, 0, stream>>>(W_proj, wpT, Cc, Cc);

  qkv_gemm_kernel<<<(Mrows / 64) * (N3 / 64) / 4, 128, 0, stream>>>(
      xbf, waT, b_attn, qw, kw, vT);

  size_t shbytes = (size_t)16 * LDSS * sizeof(float);          // 132 KB
  hipFuncSetAttribute((const void*)attn_kernel,
                      hipFuncAttributeMaxDynamicSharedMemorySize, (int)shbytes);
  attn_kernel<<<Bb * Hh * (Tt / 16), 128, shbytes, stream>>>(qw, kw, vT, att, yh);

  proj_gemm_kernel<<<(Mrows / 64) * (Cc / 64) / 4, 128, 0, stream>>>(
      yh, wpT, b_proj, yout);
}
